// MultiHeadAttention_9783935500754
// MI455X (gfx1250) — matrix operand order repaired and checked by reading, not compiled, not run on hardware
//
#include <hip/hip_runtime.h>
#include <hip/hip_bf16.h>
#include <math.h>

// ---------------------------------------------------------------------------
// MHA (causal) for B=4, T=2048, C=1024, H=16, DH=64 on gfx1250 (MI455X).
// GEMMs via v_wmma_f32_16x16x32_bf16 (wave32). K/V streamed into LDS by the
// Tensor Data Mover (tensor_load_to_lds + s_wait_tensorcnt), double-buffered.
// ---------------------------------------------------------------------------

typedef __attribute__((ext_vector_type(16))) __bf16 v16bf;
typedef __attribute__((ext_vector_type(8)))  float  v8f;
typedef __attribute__((ext_vector_type(4)))  unsigned int uvec4;
typedef __attribute__((ext_vector_type(8)))  int ivec8;
typedef __attribute__((ext_vector_type(4)))  int ivec4;

#define B_  4
#define T_  2048
#define C_  1024
#define H_  16
#define DH_ 64

struct alignas(16) U4 { unsigned int x, y, z, w; };

union ABTile {
  v16bf v;
  unsigned short s[16];
  U4 q[2];
};

__device__ __forceinline__ unsigned short f2bf_raw(float f) {
  union { float f; unsigned u; } a; a.f = f;
  unsigned r = a.u + 0x7FFFu + ((a.u >> 16) & 1u);  // round-to-nearest-even
  return (unsigned short)(r >> 16);
}
__device__ __forceinline__ unsigned pack2bf(float lo, float hi) {
  return (unsigned)f2bf_raw(lo) | ((unsigned)f2bf_raw(hi) << 16);
}

__device__ __forceinline__ v8f wmma_bf16(v16bf a, v16bf b, v8f c) {
  return __builtin_amdgcn_wmma_f32_16x16x32_bf16(
      false, a, false, b, (short)0, c, false, false);
}

// ---------------------------------------------------------------------------
// TDM: async 2D tile load Global -> LDS (Tensor DMA Descriptor per ISA ch.8).
// Rows of tile_d0 elements (2B each), tile_d1 rows, row stride stride0 elems.
// Tile is packed contiguously in LDS at lds_addr.
// ---------------------------------------------------------------------------
__device__ __forceinline__ void tdm_load_2d(unsigned lds_addr,
                                            const void* gptr,
                                            unsigned tensor_d0, unsigned tensor_d1,
                                            unsigned tile_d0, unsigned tile_d1,
                                            unsigned long long stride0) {
  const unsigned long long ga = (unsigned long long)(size_t)gptr;
  uvec4 g0;
  g0[0] = 1u;                                      // count=1, user descriptor
  g0[1] = lds_addr;                                // LDS byte address
  g0[2] = (unsigned)ga;                            // global_addr[31:0]
  g0[3] = (unsigned)((ga >> 32) & 0x01FFFFFFull)   // global_addr[56:32]
          | 0x80000000u;                           // type=2 ("image")
  ivec8 g1;
  g1[0] = (int)(1u << 16);                         // data_size=1 -> 2 bytes
  g1[1] = (int)((tensor_d0 & 0xFFFFu) << 16);      // tensor_dim0[15:0]
  g1[2] = (int)((tensor_d0 >> 16) | ((tensor_d1 & 0xFFFFu) << 16));
  g1[3] = (int)((tensor_d1 >> 16) | (tile_d0 << 16));   // tile_dim0
  g1[4] = (int)(tile_d1 & 0xFFFFu);                // tile_dim1 (tile_dim2=0)
  g1[5] = (int)(unsigned)(stride0 & 0xFFFFFFFFu);  // tensor_dim0_stride lo
  g1[6] = (int)(unsigned)((stride0 >> 32) & 0xFFFFu);
  g1[7] = 0;
  ivec4 g2 = {0, 0, 0, 0};
  ivec4 g3 = {0, 0, 0, 0};
#if defined(__clang_major__) && (__clang_major__ >= 23)
  ivec8 g4 = {0, 0, 0, 0, 0, 0, 0, 0};
  __builtin_amdgcn_tensor_load_to_lds(g0, g1, g2, g3, g4, 0);
#else
  __builtin_amdgcn_tensor_load_to_lds(g0, g1, g2, g3, 0);
#endif
}

// ---------------------------------------------------------------------------
// Prep A: fp32 -> bf16 elementwise, 8 elems/thread, packed b128 stores.
// ---------------------------------------------------------------------------
__global__ __launch_bounds__(256)
void cvt_bf16_kernel(const float* __restrict__ src,
                     unsigned short* __restrict__ dst, int n) {
  const int i = (blockIdx.x * 256 + threadIdx.x) * 8;
  if (i >= n) return;
  U4 o;
  o.x = pack2bf(src[i + 0], src[i + 1]);
  o.y = pack2bf(src[i + 2], src[i + 3]);
  o.z = pack2bf(src[i + 4], src[i + 5]);
  o.w = pack2bf(src[i + 6], src[i + 7]);
  *(U4*)(dst + i) = o;
}

// ---------------------------------------------------------------------------
// Prep B: W[3][H][C][DH] fp32 -> Wt[3][H][DH][C] bf16 (transpose C<->DH).
// ---------------------------------------------------------------------------
__global__ __launch_bounds__(256)
void cvt_w_transpose_kernel(const float* __restrict__ Wq,
                            const float* __restrict__ Wk,
                            const float* __restrict__ Wv,
                            unsigned short* __restrict__ Wt) {
  int idx = blockIdx.x * 256 + threadIdx.x;       // over 3*H*DH*C/2
  const int c2 = idx % (C_ / 2);  idx /= (C_ / 2);
  const int d  = idx % DH_;       idx /= DH_;
  const int h  = idx % H_;        idx /= H_;
  const int m  = idx;                              // 0..2
  if (m >= 3) return;
  const float* W = (m == 0) ? Wq : (m == 1) ? Wk : Wv;
  const float v0 = W[(size_t)(h * C_ + 2 * c2 + 0) * DH_ + d];
  const float v1 = W[(size_t)(h * C_ + 2 * c2 + 1) * DH_ + d];
  ((unsigned int*)Wt)[(((size_t)m * H_ + h) * DH_ + d) * (C_ / 2) + c2] =
      pack2bf(v0, v1);
}

// ---------------------------------------------------------------------------
// Kernel 1: q/k/v = x @ W[h].  q,k stored [B,H,T,DH]; v stored [B,H,DH,T].
// ---------------------------------------------------------------------------
__global__ __launch_bounds__(256)
void qkv_proj_kernel(const unsigned short* __restrict__ xb,   // bf16 [B*T, C]
                     const unsigned short* __restrict__ Wt,   // bf16 [3,H,DH,C]
                     unsigned short* __restrict__ qkv) {
  const int lane = threadIdx.x & 31;
  const int wave = threadIdx.x >> 5;
  const int hi   = lane >> 4;
  const int lo   = lane & 15;

  const int rowtile = blockIdx.x * 8 + wave;   // 0..511 over B*T/16
  const int m = blockIdx.y / H_;               // 0=q 1=k 2=v
  const int h = blockIdx.y % H_;

  const unsigned short* Wm = Wt + ((size_t)m * H_ + h) * DH_ * C_;  // [DH][C]
  unsigned short* out = qkv + (size_t)m * ((size_t)B_ * H_ * T_ * DH_);

  const int row = rowtile * 16 + lo;
  const unsigned short* xrow = xb + (size_t)row * C_;

  v8f acc[4] = {};

  for (int kk = 0; kk < C_ / 32; ++kk) {
    const int k0 = kk * 32;
    __builtin_prefetch(xrow + k0 + 256, 0, 1);

    ABTile a;
    a.q[0] = *(const U4*)(xrow + k0 + hi * 8);
    a.q[1] = *(const U4*)(xrow + k0 + 16 + hi * 8);

#pragma unroll
    for (int nt = 0; nt < 4; ++nt) {
      const unsigned short* wrow = Wm + (size_t)(nt * 16 + lo) * C_ + k0;
      ABTile b;
      b.q[0] = *(const U4*)(wrow + hi * 8);
      b.q[1] = *(const U4*)(wrow + 16 + hi * 8);
      acc[nt] = wmma_bf16(a.v, b.v, acc[nt]);
    }
  }

#pragma unroll
  for (int nt = 0; nt < 4; ++nt) {
#pragma unroll
    for (int r = 0; r < 8; ++r) {
      const int tglob = rowtile * 16 + r + 8 * hi;
      const int bidx  = tglob / T_;
      const int tt    = tglob % T_;
      const int col   = nt * 16 + lo;
      const unsigned short val = f2bf_raw(acc[nt][r]);
      if (m == 2) {  // V transposed: [B,H,DH,T]
        out[((size_t)(bidx * H_ + h) * DH_ + col) * T_ + tt] = val;
      } else {       // Q,K: [B,H,T,DH]
        out[((size_t)(bidx * H_ + h) * T_ + tt) * DH_ + col] = val;
      }
    }
  }
}

// ---------------------------------------------------------------------------
// Kernel 2: causal flash attention. One wave per 16-query tile per (b,h).
// K (32x64) and V (64x32) key-block tiles are TDM-prefetched into LDS with
// double buffering; compute overlaps the next block's DMA (TENSORcnt).
// ---------------------------------------------------------------------------
__global__ __launch_bounds__(32)
void flash_attn_kernel(const unsigned short* __restrict__ qb,
                       const unsigned short* __restrict__ kb,
                       const unsigned short* __restrict__ vtb,
                       unsigned short* __restrict__ attn) {
  __shared__ alignas(16) unsigned short Kb[2][32][64];  // [buf][key][d]
  __shared__ alignas(16) unsigned short Vb[2][64][32];  // [buf][d][key]
  __shared__ alignas(16) unsigned short Pl[16][32];

  const int lane = threadIdx.x & 31;
  const int hi = lane >> 4, lo = lane & 15;
  const int q0 = blockIdx.x * 16;
  const int bh = blockIdx.y;                   // b*H + h
  const int b  = bh / H_, h = bh % H_;

  const unsigned short* Q  = qb  + (size_t)bh * T_ * DH_;
  const unsigned short* K  = kb  + (size_t)bh * T_ * DH_;
  const unsigned short* Vt = vtb + (size_t)bh * DH_ * T_;   // [DH][T]

  // Preload Q as two A-tiles (16x32 each, covering DH=64)
  ABTile aq[2];
  {
    const unsigned short* qrow = Q + (size_t)(q0 + lo) * DH_;
#pragma unroll
    for (int kt = 0; kt < 2; ++kt) {
      aq[kt].q[0] = *(const U4*)(qrow + kt * 32 + hi * 8);
      aq[kt].q[1] = *(const U4*)(qrow + kt * 32 + 16 + hi * 8);
    }
  }

  v8f O[4] = {};
  float mrow[8], lrow[8];
#pragma unroll
  for (int r = 0; r < 8; ++r) { mrow[r] = -1e30f; lrow[r] = 0.f; }

  const int nkb = (q0 + 16 + 31) / 32;         // causal key-block bound

  // TDM issue for one 32-key block: K tile 32 rows x 64, V tile 64 rows x 32.
  auto issue = [&](int blk) {
    const int s0 = blk * 32;
    const int bi = blk & 1;
    tdm_load_2d((unsigned)(size_t)(&Kb[bi][0][0]),
                K + (size_t)s0 * DH_,
                /*tensor_d0=*/64u, /*tensor_d1=*/32u,
                /*tile_d0=*/64u, /*tile_d1=*/32u, /*stride0=*/(unsigned long long)DH_);
    tdm_load_2d((unsigned)(size_t)(&Vb[bi][0][0]),
                Vt + s0,
                /*tensor_d0=*/32u, /*tensor_d1=*/64u,
                /*tile_d0=*/32u, /*tile_d1=*/64u, /*stride0=*/(unsigned long long)T_);
  };

  issue(0);

  for (int kbk = 0; kbk < nkb; ++kbk) {
    const int s0 = kbk * 32;
    const int cur = kbk & 1;

    if (kbk + 1 < nkb) {
      issue(kbk + 1);
      __builtin_amdgcn_s_wait_tensorcnt(2);  // current block's 2 tiles done
    } else {
      __builtin_amdgcn_s_wait_tensorcnt(0);
    }
    __asm__ __volatile__("" ::: "memory");

    // S = Q K^T  (K tile rows contiguous in LDS -> ds b128 loads)
    v8f S[2];
#pragma unroll
    for (int nt = 0; nt < 2; ++nt) {
      v8f s = {};
#pragma unroll
      for (int kt = 0; kt < 2; ++kt) {
        const unsigned short* krow = &Kb[cur][nt * 16 + lo][kt * 32];
        ABTile bK;
        bK.q[0] = *(const U4*)(krow + hi * 8);
        bK.q[1] = *(const U4*)(krow + 16 + hi * 8);
        s = wmma_bf16(aq[kt].v, bK.v, s);
      }
      S[nt] = s;
    }

    // scale + causal mask (C layout: VGPR r -> row r + 8*hi; lane lo -> col)
#pragma unroll
    for (int nt = 0; nt < 2; ++nt) {
      const int kidx = s0 + nt * 16 + lo;
#pragma unroll
      for (int r = 0; r < 8; ++r) {
        const int qidx = q0 + r + 8 * hi;
        const float sv = S[nt][r] * 0.125f;    // DH^-0.5
        S[nt][r] = (kidx > qidx) ? -1e30f : sv;
      }
    }

    // online softmax update (row reductions across 16-lane halves)
#pragma unroll
    for (int r = 0; r < 8; ++r) {
      float mn = fmaxf(S[0][r], S[1][r]);
#pragma unroll
      for (int msk = 8; msk >= 1; msk >>= 1)
        mn = fmaxf(mn, __shfl_xor(mn, msk, 32));
      const float mold = mrow[r];
      const float mnew = fmaxf(mold, mn);
      const float fs = __expf(mold - mnew);
      const float p0 = __expf(S[0][r] - mnew);
      const float p1 = __expf(S[1][r] - mnew);
      S[0][r] = p0; S[1][r] = p1;
      float rs = p0 + p1;
#pragma unroll
      for (int msk = 8; msk >= 1; msk >>= 1)
        rs += __shfl_xor(rs, msk, 32);
      lrow[r] = lrow[r] * fs + rs;
      mrow[r] = mnew;
#pragma unroll
      for (int nt = 0; nt < 4; ++nt) O[nt][r] *= fs;
    }

    // P tile (C layout) -> LDS -> A layout (LDS ops are in-order per wave)
#pragma unroll
    for (int nt = 0; nt < 2; ++nt)
#pragma unroll
      for (int r = 0; r < 8; ++r)
        Pl[r + 8 * hi][nt * 16 + lo] = f2bf_raw(S[nt][r]);

    ABTile pa;
    pa.q[0] = *(const U4*)(&Pl[lo][hi * 8]);
    pa.q[1] = *(const U4*)(&Pl[lo][16 + hi * 8]);

    // O += P (16x32) x V (32x64); V tile rows contiguous in keys (LDS)
#pragma unroll
    for (int nt = 0; nt < 4; ++nt) {
      const unsigned short* vrow = &Vb[cur][nt * 16 + lo][0];
      ABTile bV;
      bV.q[0] = *(const U4*)(vrow + hi * 8);
      bV.q[1] = *(const U4*)(vrow + 16 + hi * 8);
      O[nt] = wmma_bf16(pa.v, bV.v, O[nt]);
    }
  }

  // normalize and store bf16 into attn[B*T, C] (head-concat layout)
#pragma unroll
  for (int nt = 0; nt < 4; ++nt) {
#pragma unroll
    for (int r = 0; r < 8; ++r) {
      const int row = b * T_ + q0 + r + 8 * hi;
      const int col = h * DH_ + nt * 16 + lo;
      attn[(size_t)row * C_ + col] = f2bf_raw(O[nt][r] / lrow[r]);
    }
  }
}

// ---------------------------------------------------------------------------
// Kernel 3: y = attn @ Wo^T + bo  (fp32 out). Wo rows contiguous in K.
// ---------------------------------------------------------------------------
__global__ __launch_bounds__(128)
void out_proj_kernel(const unsigned short* __restrict__ attn,   // bf16
                     const unsigned short* __restrict__ Wob,    // bf16 [C,C]
                     const float* __restrict__ bo,
                     float* __restrict__ y) {
  const int lane = threadIdx.x & 31;
  const int wave = threadIdx.x >> 5;
  const int hi = lane >> 4, lo = lane & 15;
  const int rowtile = blockIdx.x * 4 + wave;   // 0..511
  const int j0 = blockIdx.y * 64;

  const int row = rowtile * 16 + lo;
  const unsigned short* arow = attn + (size_t)row * C_;

  v8f acc[4] = {};
  for (int kk = 0; kk < C_ / 32; ++kk) {
    const int c0 = kk * 32;
    __builtin_prefetch(arow + c0 + 256, 0, 1);

    ABTile a;
    a.q[0] = *(const U4*)(arow + c0 + hi * 8);
    a.q[1] = *(const U4*)(arow + c0 + 16 + hi * 8);

#pragma unroll
    for (int nt = 0; nt < 4; ++nt) {
      const unsigned short* wrow =
          Wob + (size_t)(j0 + nt * 16 + lo) * C_ + c0;
      ABTile bw;
      bw.q[0] = *(const U4*)(wrow + hi * 8);
      bw.q[1] = *(const U4*)(wrow + 16 + hi * 8);
      acc[nt] = wmma_bf16(a.v, bw.v, acc[nt]);
    }
  }

#pragma unroll
  for (int nt = 0; nt < 4; ++nt) {
    const float bias = bo[j0 + nt * 16 + lo];
#pragma unroll
    for (int r = 0; r < 8; ++r) {
      const int rr = rowtile * 16 + r + 8 * hi;
      y[(size_t)rr * C_ + j0 + nt * 16 + lo] = acc[nt][r] + bias;
    }
  }
}

// ---------------------------------------------------------------------------
extern "C" void kernel_launch(void* const* d_in, const int* in_sizes, int n_in,
                              void* d_out, int out_size, void* d_ws, size_t ws_size,
                              hipStream_t stream) {
  (void)in_sizes; (void)n_in; (void)out_size; (void)ws_size;

  const float* x  = (const float*)d_in[0];
  const float* Wq = (const float*)d_in[1];
  const float* Wk = (const float*)d_in[2];
  const float* Wv = (const float*)d_in[3];
  const float* Wo = (const float*)d_in[4];
  const float* bo = (const float*)d_in[5];
  float* y = (float*)d_out;

  // workspace layout (bf16 elements)
  const size_t n_x    = (size_t)B_ * T_ * C_;          //  8,388,608 (16 MiB)
  const size_t n_wt   = (size_t)3 * H_ * DH_ * C_;     //  3,145,728 ( 6 MiB)
  const size_t n_wo   = (size_t)C_ * C_;               //  1,048,576 ( 2 MiB)
  const size_t n_mat  = (size_t)B_ * H_ * T_ * DH_;    //  8,388,608 (16 MiB)

  unsigned short* xb   = (unsigned short*)d_ws;
  unsigned short* Wt   = xb + n_x;
  unsigned short* Wob  = Wt + n_wt;
  unsigned short* qkv  = Wob + n_wo;                   // 3 matrices
  unsigned short* attn = qkv + 3 * n_mat;

  // --- prep: convert to bf16 (x, Wo) and convert+transpose W ---
  cvt_bf16_kernel<<<(int)(n_x / 8 / 256), 256, 0, stream>>>(x, xb, (int)n_x);
  cvt_bf16_kernel<<<(int)(n_wo / 8 / 256), 256, 0, stream>>>(Wo, Wob, (int)n_wo);
  cvt_w_transpose_kernel<<<(int)(n_wt / 2 / 256), 256, 0, stream>>>(Wq, Wk, Wv, Wt);

  // --- QKV projection ---
  {
    dim3 grid(64, 3 * H_);          // 64 blocks x 8 waves = 512 row tiles
    qkv_proj_kernel<<<grid, 256, 0, stream>>>(xb, Wt, qkv);
  }
  // --- flash attention ---
  {
    dim3 grid(T_ / 16, B_ * H_);    // one wave-block per 16-query tile
    flash_attn_kernel<<<grid, 32, 0, stream>>>(qkv, qkv + n_mat,
                                               qkv + 2 * n_mat, attn);
  }
  // --- output projection ---
  {
    dim3 grid(128, C_ / 64);        // 128 blocks x 4 waves = 512 row tiles
    out_proj_kernel<<<grid, 128, 0, stream>>>(attn, Wob, bo, y);
  }
}
